// NmnistMlp_30236569764071
// MI455X (gfx1250) — matrix operand order repaired and checked by reading, not compiled, not run on hardware
//
#include <hip/hip_runtime.h>
#include <hip/hip_bf16.h>

typedef __attribute__((ext_vector_type(16))) _Float16 v16h;
typedef __attribute__((ext_vector_type(8)))  float    v8f;

#define TT   300
#define NBAT 64
#define NN   (TT * NBAT)      // 19200  (b,t) folded GEMM-N
#define KLEN 64
#define THETA_F 10.0f
#define LDA  40               // LDS row stride in halves (32 data + 8 pad)

#ifndef __has_builtin
#define __has_builtin(x) 0
#endif
#if __has_builtin(__builtin_amdgcn_global_load_async_to_lds_b128) && \
    __has_builtin(__builtin_amdgcn_s_wait_asynccnt)
#define USE_ASYNC_LDS 1
#else
#define USE_ASYNC_LDS 0
#endif

#if USE_ASYNC_LDS
typedef int v4i_ __attribute__((vector_size(16)));
typedef __attribute__((address_space(1))) v4i_ g_v4i;   // global (prints as __device__)
typedef __attribute__((address_space(3))) v4i_ l_v4i;   // LDS
// one 16-byte async global->LDS transfer per lane
__device__ __forceinline__ void async_b128(const _Float16* g, _Float16* l) {
  __builtin_amdgcn_global_load_async_to_lds_b128(
      (g_v4i*)(unsigned long long)g,
      (l_v4i*)(unsigned int)(unsigned long long)l,   // flat->LDS = addr[31:0]
      0, 0);
}
#endif

// ---------------- weight conversion: f32 [M][K] -> f16 [Mpad][Kpad], zero padded
__global__ void cvt_w_kernel(const float* __restrict__ W, _Float16* __restrict__ Wh,
                             int M, int K, int Kpad, int total) {
  int i = blockIdx.x * 256 + threadIdx.x;
  if (i >= total) return;
  int m = i / Kpad, k = i - m * Kpad;
  float v = (m < M && k < K) ? W[(size_t)m * K + k] : 0.0f;
  Wh[i] = (_Float16)v;
}

// ---------------- input transpose: f32 [B][K][T] -> f16 S_t[n][Kpad], n = b*TT+t
__global__ void cvt_s0_kernel(const float* __restrict__ S, _Float16* __restrict__ St,
                              int K, int Kpad) {
  __shared__ _Float16 tile[32][TT];
  const int c0 = blockIdx.x * 32;
  const int b  = blockIdx.y;
  for (int j = threadIdx.x; j < 32 * TT; j += 256) {
    const int cl = j / TT, t = j - cl * TT;      // consecutive j -> consecutive t
    const int c = c0 + cl;
    const float v = (c < K) ? S[((size_t)b * K + c) * TT + t] : 0.0f;
    tile[cl][t] = (_Float16)v;
  }
  __syncthreads();
  for (int j = threadIdx.x; j < 32 * TT; j += 256) {
    const int t = j >> 5, cl = j & 31;           // consecutive j -> consecutive c
    St[((size_t)b * TT + t) * Kpad + c0 + cl] = tile[cl][t];
  }
}

// ---------------- fragment loaders (contiguous k = 16*g + s per lane half, same for A and B) ----------------
__device__ __forceinline__ v16h load_fragA(const _Float16* p) {
  union { v16h v; uint2 q[4]; } u;
  const uint2* a = (const uint2*)p;        // K: +0..7
  const uint2* b = (const uint2*)(p + 8);  // K: +8..15
  u.q[0] = a[0]; u.q[1] = a[1]; u.q[2] = b[0]; u.q[3] = b[1];
  return u.v;
}
__device__ __forceinline__ v16h load_fragB(const _Float16* p) {
  union { v16h v; uint2 q[4]; } u;
  const uint2* a = (const uint2*)p;
  u.q[0] = a[0]; u.q[1] = a[1]; u.q[2] = a[2]; u.q[3] = a[3];
  return u.v;
}

// ---------------- GEMM: Z[Mpad][NN] = Wh[Mpad][Kpad] * S_t[NN][Kpad]^T ----------------
// Block tile 64(M) x 128(N), 128 threads (4 waves), each wave 32x64 => 8 WMMA/chunk.
// Double-buffered LDS; async global->LDS copies when the toolchain exposes them.
__global__ void gemm_kernel(const _Float16* __restrict__ St,
                            const _Float16* __restrict__ Wh,
                            float* __restrict__ Z,
                            int Kpad) {
  __shared__ _Float16 As[2][64 * LDA];
  __shared__ _Float16 Bs[2][128 * LDA];

  const int tid  = threadIdx.x;       // 0..127 (4 waves)
  const int lane = tid & 31;
  const int wave = tid >> 5;
  const int mw   = (wave >> 1) * 32;  // wave M origin (0/32)
  const int nw   = (wave & 1) * 64;   // wave N origin (0/64)
  const int nBlock = blockIdx.x * 128;
  const int mBlock = blockIdx.y * 64;

  const int g = lane >> 4;            // half-wave group
  const int r = lane & 15;

  v8f acc[2][4];
  #pragma unroll
  for (int i = 0; i < 2; ++i)
    #pragma unroll
    for (int j = 0; j < 4; ++j) acc[i][j] = (v8f){};

  // staging roles: A: 2 threads/row (64 rows, 16 halves each); B: 1 thread/row (128 rows, 32 halves)
  const int aRow  = tid >> 1;
  const int aColH = (tid & 1) * 16;
  const _Float16* aGlob = Wh + (size_t)(mBlock + aRow) * Kpad + aColH;
  const _Float16* bGlob = St + (size_t)(nBlock + tid) * Kpad;

#if USE_ASYNC_LDS
  auto stage = [&](int kc, int buf) {
    _Float16* al = &As[buf][aRow * LDA + aColH];
    async_b128(aGlob + kc,     al);
    async_b128(aGlob + kc + 8, al + 8);
    _Float16* bl = &Bs[buf][tid * LDA];
    async_b128(bGlob + kc,      bl);
    async_b128(bGlob + kc + 8,  bl + 8);
    async_b128(bGlob + kc + 16, bl + 16);
    async_b128(bGlob + kc + 24, bl + 24);
  };
  stage(0, 0);
  __builtin_amdgcn_s_wait_asynccnt(0);
  __syncthreads();
  int cur = 0;
  for (int kc = 0; kc < Kpad; kc += 32) {
    if (kc + 32 < Kpad) stage(kc + 32, cur ^ 1);  // overlap copy with compute

    const _Float16* Asb = As[cur];
    const _Float16* Bsb = Bs[cur];
    v16h a0 = load_fragA(&Asb[(mw +      r) * LDA + g * 16]);
    v16h a1 = load_fragA(&Asb[(mw + 16 + r) * LDA + g * 16]);
    #pragma unroll
    for (int tj = 0; tj < 4; ++tj) {
      v16h b = load_fragB(&Bsb[(nw + tj * 16 + r) * LDA + g * 16]);
      acc[0][tj] = __builtin_amdgcn_wmma_f32_16x16x32_f16(false, a0, false, b, (short)0, acc[0][tj], false, false);
      acc[1][tj] = __builtin_amdgcn_wmma_f32_16x16x32_f16(false, a1, false, b, (short)0, acc[1][tj], false, false);
    }
    __builtin_amdgcn_s_wait_asynccnt(0);
    __syncthreads();
    cur ^= 1;
  }
#else
  for (int kc = 0; kc < Kpad; kc += 32) {
    {
      const uint4* gs = (const uint4*)(aGlob + kc);
      uint4* dst = (uint4*)(&As[0][aRow * LDA + aColH]);
      dst[0] = gs[0];
      dst[1] = gs[1];
    }
    {
      const uint4* gs = (const uint4*)(bGlob + kc);
      uint4* dst = (uint4*)(&Bs[0][tid * LDA]);
      dst[0] = gs[0]; dst[1] = gs[1]; dst[2] = gs[2]; dst[3] = gs[3];
    }
    if (kc + 32 < Kpad) {
      __builtin_prefetch(aGlob + kc + 32, 0, 0);
      __builtin_prefetch(bGlob + kc + 32, 0, 0);
    }
    __syncthreads();
    v16h a0 = load_fragA(&As[0][(mw +      r) * LDA + g * 16]);
    v16h a1 = load_fragA(&As[0][(mw + 16 + r) * LDA + g * 16]);
    #pragma unroll
    for (int tj = 0; tj < 4; ++tj) {
      v16h b = load_fragB(&Bs[0][(nw + tj * 16 + r) * LDA + g * 16]);
      acc[0][tj] = __builtin_amdgcn_wmma_f32_16x16x32_f16(false, a0, false, b, (short)0, acc[0][tj], false, false);
      acc[1][tj] = __builtin_amdgcn_wmma_f32_16x16x32_f16(false, a1, false, b, (short)0, acc[1][tj], false, false);
    }
    __syncthreads();
  }
#endif

  // ---- epilogue: C/D layout: VGPR rr -> M = rr + (lane>=16 ? 8 : 0), N = lane&15
  const int mo = mBlock + mw + (g ? 8 : 0);
  const int no = nBlock + nw + r;
  #pragma unroll
  for (int ti = 0; ti < 2; ++ti)
    #pragma unroll
    for (int tj = 0; tj < 4; ++tj) {
      float* zr = Z + (size_t)(mo + ti * 16) * NN + no + tj * 16;
      #pragma unroll
      for (int rr = 0; rr < 8; ++rr)
        zr[(size_t)rr * NN] = acc[ti][tj][rr];
    }
}

// ---------------- PSP: 64-tap causal alpha FIR along t, in-place, block per (o,b)
__global__ void psp_kernel(float* __restrict__ Z) {
  __shared__ float xs[KLEN + TT];
  __shared__ float taps[KLEN];
  const int o = blockIdx.x / NBAT;
  const int b = blockIdx.x - o * NBAT;
  const int tid = threadIdx.x;    // 128

  float* zr = Z + (size_t)o * NN + b * TT;
  for (int i = tid; i < KLEN + TT; i += 128)
    xs[i] = (i < KLEN) ? 0.0f : zr[i - KLEN];
  if (tid < KLEN) {
    const float tm = (float)(tid + 1);
    taps[tid] = (tm / 8.0f) * __expf(1.0f - tm / 8.0f);
  }
  __syncthreads();

  for (int t = tid; t < TT; t += 128) {
    float acc = 0.0f;
    #pragma unroll 8
    for (int m = 0; m < KLEN; ++m)
      acc += taps[m] * xs[KLEN + t - m];
    zr[t] = acc;
  }
}

// ---------------- spike dynamics: bitmask refractory scan; transposed f16 out
__global__ void spike_mid_kernel(const float* __restrict__ Z,
                                 _Float16* __restrict__ St,
                                 int Mrows, int Kpad) {
  __shared__ float    zt[64][65];
  __shared__ _Float16 st[64][64];
  __shared__ float    refk[KLEN];
  const int tid = threadIdx.x;                 // 64
  {
    const float tm = (float)(tid + 1);
    refk[tid] = -2.0f * 10.0f * (tm / 8.0f) * __expf(1.0f - tm / 8.0f);
  }
  const int grpPerB = Mrows >> 6;
  const int b  = blockIdx.x / grpPerB;
  const int og = (blockIdx.x - b * grpPerB) * 64;
  __syncthreads();

  unsigned long long mask = 0ull;              // bit p => spike at lag p+1
  for (int tc = 0; tc < TT; tc += 64) {
    const int len = (TT - tc < 64) ? (TT - tc) : 64;
    for (int j = tid; j < 64 * len; j += 64) {
      const int rr = j / len, tl = j - rr * len;   // coalesced along t
      zt[rr][tl] = Z[(size_t)(og + rr) * NN + b * TT + tc + tl];
    }
    __syncthreads();
    for (int tl = 0; tl < len; ++tl) {
      float r = 0.0f;
      unsigned long long m = mask;
      while (m) {
        const int p = __ffsll((long long)m) - 1;
        r += refk[p];
        m &= m - 1;
      }
      const float v = zt[tid][tl] + r - THETA_F;
      const int s = (v >= 0.0f) ? 1 : 0;
      st[tl][tid] = (_Float16)(float)s;
      mask = (mask << 1) | (unsigned long long)s;
    }
    __syncthreads();
    for (int j = tid; j < len * 64; j += 64) {
      const int tl = j >> 6, c = j & 63;           // coalesced along k
      St[((size_t)b * TT + tc + tl) * Kpad + og + c] = st[tl][c];
    }
    __syncthreads();
  }
}

__global__ void spike_last_kernel(const float* __restrict__ U,
                                  float* __restrict__ out, int Mrows) {
  __shared__ float refk[KLEN];
  if (threadIdx.x < KLEN) {
    const float tm = (float)(threadIdx.x + 1);
    refk[threadIdx.x] = -2.0f * 10.0f * (tm / 8.0f) * __expf(1.0f - tm / 8.0f);
  }
  __syncthreads();

  const int gid = blockIdx.x * 64 + threadIdx.x;
  if (gid >= Mrows * NBAT) return;
  const int o = gid / NBAT, b = gid - o * NBAT;
  const float* ur = U + (size_t)o * NN + b * TT;
  float*       orow = out + ((size_t)b * Mrows + o) * TT;  // [B, 10, T]

  unsigned long long mask = 0ull;
  for (int t = 0; t < TT; ++t) {
    float r = 0.0f;
    unsigned long long m = mask;
    while (m) {
      const int p = __ffsll((long long)m) - 1;
      r += refk[p];
      m &= m - 1;
    }
    const float v = ur[t] + r - THETA_F;
    const int s = (v >= 0.0f) ? 1 : 0;
    orow[t] = (float)s;
    mask = (mask << 1) | (unsigned long long)s;
  }
}

// ---------------- launcher ----------------
extern "C" void kernel_launch(void* const* d_in, const int* in_sizes, int n_in,
                              void* d_out, int out_size, void* d_ws, size_t ws_size,
                              hipStream_t stream) {
  const float* s_in = (const float*)d_in[0];   // [64, 2312, 300]
  const float* W1   = (const float*)d_in[1];   // [512, 2312]
  const float* W2   = (const float*)d_in[2];   // [512, 512]
  const float* W3   = (const float*)d_in[3];   // [10, 512]

  const int K1 = 2312, K1P = 2336;             // pad K to /32
  const int M12 = 512, M3 = 10, M3P = 64;
  const int K2 = 512;

  char* ws = (char*)d_ws;
  size_t off = 0;
  _Float16* W1h = (_Float16*)(ws + off); off += (size_t)M12 * K1P * 2;  off = (off + 255) & ~(size_t)255;
  _Float16* W2h = (_Float16*)(ws + off); off += (size_t)M12 * K2 * 2;   off = (off + 255) & ~(size_t)255;
  _Float16* W3h = (_Float16*)(ws + off); off += (size_t)M3P * K2 * 2;   off = (off + 255) & ~(size_t)255;
  _Float16* S0  = (_Float16*)(ws + off); off += (size_t)K1P * NN * 2;   off = (off + 255) & ~(size_t)255;
  float*    Z   = (float*)(ws + off);    off += (size_t)M12 * NN * 4;   off = (off + 255) & ~(size_t)255;
  _Float16* S1  = (_Float16*)(ws + off); off += (size_t)M12 * NN * 2;
  _Float16* S2  = S0;                          // S0 dead after layer-1 GEMM; reuse

  {
    int t1 = M12 * K1P;
    cvt_w_kernel<<<(t1 + 255) / 256, 256, 0, stream>>>(W1, W1h, M12, K1, K1P, t1);
    int t2 = M12 * K2;
    cvt_w_kernel<<<(t2 + 255) / 256, 256, 0, stream>>>(W2, W2h, M12, K2, K2, t2);
    int t3 = M3P * K2;
    cvt_w_kernel<<<(t3 + 255) / 256, 256, 0, stream>>>(W3, W3h, M3, K2, K2, t3);
    cvt_s0_kernel<<<dim3(K1P / 32, NBAT), 256, 0, stream>>>(s_in, S0, K1, K1P);
  }

  const dim3 gemmBlk(128);
  const dim3 gemmGrd12(NN / 128, M12 / 64);   // (150, 8)
  const dim3 gemmGrd3 (NN / 128, 1);          // (150, 1)
  const int spikeGrd = NBAT * (M12 / 64);     // 512 blocks of 64 threads

  // ---- layer 1 ----
  gemm_kernel<<<gemmGrd12, gemmBlk, 0, stream>>>(S0, W1h, Z, K1P);
  psp_kernel<<<M12 * NBAT, 128, 0, stream>>>(Z);
  spike_mid_kernel<<<spikeGrd, 64, 0, stream>>>(Z, S1, M12, K2);

  // ---- layer 2 ----
  gemm_kernel<<<gemmGrd12, gemmBlk, 0, stream>>>(S1, W2h, Z, K2);
  psp_kernel<<<M12 * NBAT, 128, 0, stream>>>(Z);
  spike_mid_kernel<<<spikeGrd, 64, 0, stream>>>(Z, S2, M12, K2);

  // ---- layer 3 ----
  gemm_kernel<<<gemmGrd3, gemmBlk, 0, stream>>>(S2, W3h, Z, K2);
  psp_kernel<<<M3 * NBAT, 128, 0, stream>>>(Z);
  spike_last_kernel<<<(M3 * NBAT) / 64, 64, 0, stream>>>(Z, (float*)d_out, M3);
}
